// GRADE_GC_22995254903249
// MI455X (gfx1250) — matrix operand order repaired and checked by reading, not compiled, not run on hardware
//
#include <hip/hip_runtime.h>

// ---------------- problem constants (from reference) ----------------
#define N_NODES 50000
#define N_EDGES 800000
#define N_GRAPH 256
#define DIM     128
#define NCLS    10
#define MTILES  (N_NODES / 16)   // 3125 exactly

typedef _Float16 v8h  __attribute__((ext_vector_type(8)));
typedef _Float16 v16h __attribute__((ext_vector_type(16)));
typedef float    v8f  __attribute__((ext_vector_type(8)));

// ---------------- utility kernels ----------------
__global__ void k_fill(float* p, float v, int n) {
    int i = blockIdx.x * blockDim.x + threadIdx.x;
    if (i < n) p[i] = v;
}

__global__ void k_deg_count(const int* __restrict__ dst, float* __restrict__ deg, int e) {
    int i = blockIdx.x * blockDim.x + threadIdx.x;
    if (i < e) atomicAdd(&deg[dst[i]], 1.0f);
}

__global__ void k_rsqrt_inplace(float* p, int n) {
    int i = blockIdx.x * blockDim.x + threadIdx.x;
    if (i < n) p[i] = rsqrtf(p[i]);   // deg >= 1 always (self-loop)
}

__global__ void k_count_batch(const int* __restrict__ batch, float* __restrict__ cnt, int n) {
    int i = blockIdx.x * blockDim.x + threadIdx.x;
    if (i < n) atomicAdd(&cnt[batch[i]], 1.0f);
}

__global__ void k_f32_to_f16(const float* __restrict__ s, _Float16* __restrict__ d, int n) {
    int i = blockIdx.x * blockDim.x + threadIdx.x;
    if (i < n) d[i] = (_Float16)s[i];
}

// W (f32, KxN row-major) -> Wt (f16, N-major: Wt[n*DIM+k] = W[k*DIM+n]).
// Done once per layer so the GEMM's LDS tile is column-contiguous.
__global__ void k_w_to_f16_t(const float* __restrict__ W, _Float16* __restrict__ Wt) {
    int i = blockIdx.x * blockDim.x + threadIdx.x;   // i = n*DIM + k
    if (i >= DIM * DIM) return;
    int n = i >> 7, k = i & 127;
    Wt[i] = (_Float16)W[k * DIM + n];
}

// ---------------- WMMA GEMM: H[N,128] = X[N,128] @ W[128,128] ----------------
// One wave computes one 16x16 output tile; 8 waves/block cover the 8 N-tiles of
// one 16-row M-tile. Wt (f16, N-major, 32KB) is staged in LDS once per block so
// B fragments are contiguous ds_load_b128 reads.
__global__ __launch_bounds__(256) void k_gemm_wmma(const _Float16* __restrict__ X,
                                                   const _Float16* __restrict__ Wt,
                                                   float* __restrict__ H) {
    __shared__ _Float16 Wlds[DIM * DIM];   // N-major: Wlds[n*DIM + k]
    {
        const v8h* sv = reinterpret_cast<const v8h*>(Wt);
        v8h* dv = reinterpret_cast<v8h*>(Wlds);
        for (int i = threadIdx.x; i < (DIM * DIM) / 8; i += 256) dv[i] = sv[i];
    }
    __syncthreads();

    const int wave   = threadIdx.x >> 5;
    const int lane   = threadIdx.x & 31;
    const int laneLo = lane & 15;
    const int laneHi = lane >> 4;
    const int mtile  = blockIdx.x;     // 0..3124
    const int ntile  = wave;           // 0..7

    const int row = mtile * 16 + laneLo;   // A-matrix row held by this lane
    const int col = ntile * 16 + laneLo;   // B/C/D column held by this lane

    v8f acc = {0.f, 0.f, 0.f, 0.f, 0.f, 0.f, 0.f, 0.f};

    #pragma unroll
    for (int kc = 0; kc < 4; ++kc) {       // K = 128 in chunks of 32
        // A fragment: 16-bit A 16x32 layout. Lanes 0-15: K = kb..kb+7 and
        // kb+16..kb+23 with kb = kc*32; lanes 16-31: same +8.
        const int kb = kc * 32 + laneHi * 8;
        v8h alo = *reinterpret_cast<const v8h*>(X + (size_t)row * DIM + kb);
        v8h ahi = *reinterpret_cast<const v8h*>(X + (size_t)row * DIM + kb + 16);
        v16h a;
        #pragma unroll
        for (int i = 0; i < 8; ++i) { a[i] = alo[i]; a[8 + i] = ahi[i]; }

        // B fragment: same K order as A. Lanes 0-15 hold K = kbb..kbb+7 and
        // kbb+16..kbb+23 of column `col`; lanes 16-31: same +8. With the N-major
        // LDS tile each 8-half run is contiguous -> two ds_load_b128.
        const int kbb = kc * 32 + laneHi * 8;
        v8h blo = *reinterpret_cast<const v8h*>(Wlds + (size_t)col * DIM + kbb);
        v8h bhi = *reinterpret_cast<const v8h*>(Wlds + (size_t)col * DIM + kbb + 16);
        v16h b;
        #pragma unroll
        for (int i = 0; i < 8; ++i) { b[i] = blo[i]; b[8 + i] = bhi[i]; }

        acc = __builtin_amdgcn_wmma_f32_16x16x32_f16(
            /*neg_a=*/false, a, /*neg_b=*/false, b,
            /*c_mod=*/(short)0, acc, /*reuse_a=*/false, /*reuse_b=*/false);
    }

    // C/D layout: VGPR r -> M = r (lanes 0-15) / 8+r (lanes 16-31), N = col.
    const int rbase = mtile * 16 + laneHi * 8;
    #pragma unroll
    for (int r = 0; r < 8; ++r)
        H[(size_t)(rbase + r) * DIM + col] = acc[r];
}

// agg[n,j] = h[n,j]*dinv[n]^2 + bias[j]   (self-loop term + bias; init before scatter)
__global__ void k_agg_init(float* __restrict__ agg, const float* __restrict__ h,
                           const float* __restrict__ dinv, const float* __restrict__ bias,
                           int n) {
    int i = blockIdx.x * blockDim.x + threadIdx.x;
    if (i >= n) return;
    int node = i >> 7, j = i & 127;
    float d = dinv[node];
    agg[i] = h[i] * d * d + bias[j];
}

// Edge scatter: agg[dst] += h[src] * dinv[src]*dinv[dst].
// One thread per (edge, 4-float chunk): 32 threads cover one 128-float row.
__global__ __launch_bounds__(256) void k_scatter(const int* __restrict__ src,
                                                 const int* __restrict__ dst,
                                                 const float* __restrict__ h,
                                                 const float* __restrict__ dinv,
                                                 float* __restrict__ agg, int e) {
    int t = blockIdx.x * blockDim.x + threadIdx.x;
    int ed = t >> 5;
    int c  = (t & 31) << 2;
    if (ed >= e) return;
    if ((t & 1023) == 0) {   // emits global_prefetch_b8 on the edge stream
        __builtin_prefetch(src + ed + 8192, 0, 1);
        __builtin_prefetch(dst + ed + 8192, 0, 1);
    }
    int s = src[ed], d = dst[ed];
    float w = dinv[s] * dinv[d];
    const float4 v = *reinterpret_cast<const float4*>(h + (size_t)s * DIM + c);
    float* pa = agg + (size_t)d * DIM + c;
    atomicAdd(pa + 0, v.x * w);
    atomicAdd(pa + 1, v.y * w);
    atomicAdd(pa + 2, v.z * w);
    atomicAdd(pa + 3, v.w * w);
}

__global__ void k_pool_sum(const float* __restrict__ agg, const int* __restrict__ batch,
                           float* __restrict__ pool, int n) {
    int i = blockIdx.x * blockDim.x + threadIdx.x;
    if (i >= n) return;
    int node = i >> 7, j = i & 127;
    atomicAdd(&pool[(size_t)batch[node] * DIM + j], agg[i]);
}

__global__ void k_pool_div(float* __restrict__ pool, const float* __restrict__ cnt, int n) {
    int i = blockIdx.x * blockDim.x + threadIdx.x;
    if (i >= n) return;
    pool[i] /= fmaxf(cnt[i >> 7], 1.0f);
}

// FC head: logits = relu(pool2 @ fcW1 + fcb1) @ fcW2 + fcb2   (one thread/graph)
__global__ void k_fc(const float* __restrict__ pool2,
                     const float* __restrict__ fcW1, const float* __restrict__ fcb1,
                     const float* __restrict__ fcW2, const float* __restrict__ fcb2,
                     float* __restrict__ logits, int g) {
    int i = blockIdx.x * blockDim.x + threadIdx.x;
    if (i >= g) return;
    float hid[16];
    #pragma unroll
    for (int j = 0; j < 16; ++j) {
        float s = fcb1[j];
        for (int k = 0; k < DIM; ++k) s += pool2[(size_t)i * DIM + k] * fcW1[k * 16 + j];
        hid[j] = fmaxf(s, 0.0f);
    }
    #pragma unroll
    for (int c = 0; c < NCLS; ++c) {
        float s = fcb2[c];
        #pragma unroll
        for (int j = 0; j < 16; ++j) s += hid[j] * fcW2[j * NCLS + c];
        logits[(size_t)i * NCLS + c] = s;
    }
}

// BCE-with-logits vs one-hot, summed into accum[0]
__global__ void k_class_loss(const float* __restrict__ logit_s, const int* __restrict__ labels,
                             float* __restrict__ accum) {
    int i = blockIdx.x * blockDim.x + threadIdx.x;
    if (i >= N_GRAPH * NCLS) return;
    int g = i / NCLS, c = i % NCLS;
    float z = logit_s[i];
    float y = (labels[g] == c) ? 1.0f : 0.0f;
    float v = fmaxf(z, 0.0f) - z * y + log1pf(expf(-fabsf(z)));
    atomicAdd(&accum[0], v);
}

// Domain discriminator branch: preds = [p1|p2|logits] @ dW + db, -log_softmax[label]
__global__ void k_domain_loss(const float* __restrict__ p1s, const float* __restrict__ p2s,
                              const float* __restrict__ ls,
                              const float* __restrict__ p1t, const float* __restrict__ p2t,
                              const float* __restrict__ lt,
                              const float* __restrict__ dW, const float* __restrict__ db,
                              float* __restrict__ accum) {
    int r = blockIdx.x * blockDim.x + threadIdx.x;
    if (r >= 2 * N_GRAPH) return;
    int g = r & (N_GRAPH - 1);
    bool tdom = (r >= N_GRAPH);
    const float* p1 = tdom ? p1t : p1s;
    const float* p2 = tdom ? p2t : p2s;
    const float* lg = tdom ? lt  : ls;
    float pr0 = db[0], pr1 = db[1];
    for (int k = 0; k < DIM; ++k) {
        float f = p1[(size_t)g * DIM + k];
        pr0 += f * dW[k * 2 + 0]; pr1 += f * dW[k * 2 + 1];
    }
    for (int k = 0; k < DIM; ++k) {
        float f = p2[(size_t)g * DIM + k];
        pr0 += f * dW[(DIM + k) * 2 + 0]; pr1 += f * dW[(DIM + k) * 2 + 1];
    }
    #pragma unroll
    for (int k = 0; k < NCLS; ++k) {
        float f = lg[(size_t)g * NCLS + k];
        pr0 += f * dW[(2 * DIM + k) * 2 + 0]; pr1 += f * dW[(2 * DIM + k) * 2 + 1];
    }
    float m   = fmaxf(pr0, pr1);
    float lse = m + logf(expf(pr0 - m) + expf(pr1 - m));
    float logp = (tdom ? pr1 : pr0) - lse;
    atomicAdd(&accum[1], -logp);
}

__global__ void k_finalize(const float* __restrict__ accum, float* __restrict__ out) {
    if (threadIdx.x == 0 && blockIdx.x == 0) {
        float cl = accum[0] / (float)(N_GRAPH * NCLS);
        float dl = accum[1] / (float)(2 * N_GRAPH);
        out[0] = cl + 0.01f * dl;
        out[1] = cl;
        out[2] = dl;
    }
}

// ---------------- host-side orchestration ----------------
static inline unsigned gdim(long long n, int b) { return (unsigned)((n + b - 1) / b); }

extern "C" void kernel_launch(void* const* d_in, const int* in_sizes, int n_in,
                              void* d_out, int out_size, void* d_ws, size_t ws_size,
                              hipStream_t stream) {
    (void)in_sizes; (void)n_in; (void)out_size; (void)ws_size;

    const float* feat_s  = (const float*)d_in[0];
    const float* feat_t  = (const float*)d_in[1];
    const int*   ei_s    = (const int*)d_in[2];
    const int*   ei_t    = (const int*)d_in[3];
    const int*   batch_s = (const int*)d_in[4];
    const int*   batch_t = (const int*)d_in[5];
    const int*   labels  = (const int*)d_in[6];
    const float* W1  = (const float*)d_in[7];  const float* b1  = (const float*)d_in[8];
    const float* W2  = (const float*)d_in[9];  const float* b2  = (const float*)d_in[10];
    const float* fcW1= (const float*)d_in[11]; const float* fcb1= (const float*)d_in[12];
    const float* fcW2= (const float*)d_in[13]; const float* fcb2= (const float*)d_in[14];
    const float* dW  = (const float*)d_in[15]; const float* db  = (const float*)d_in[16];

    // ---- carve workspace (256B aligned regions) ----
    char* p = (char*)d_ws;
    auto carve = [&](size_t bytes) -> void* {
        void* r = (void*)p;
        p += (bytes + 255) & ~(size_t)255;
        return r;
    };
    float*    dinv_s = (float*)carve(sizeof(float) * N_NODES);
    float*    dinv_t = (float*)carve(sizeof(float) * N_NODES);
    float*    cnt_s  = (float*)carve(sizeof(float) * N_GRAPH);
    float*    cnt_t  = (float*)carve(sizeof(float) * N_GRAPH);
    _Float16* xh     = (_Float16*)carve(sizeof(_Float16) * (size_t)N_NODES * DIM);
    _Float16* Wt     = (_Float16*)carve(sizeof(_Float16) * DIM * DIM);
    float*    hbuf   = (float*)carve(sizeof(float) * (size_t)N_NODES * DIM);
    float*    agg_s  = (float*)carve(sizeof(float) * (size_t)N_NODES * DIM);
    float*    agg_t  = (float*)carve(sizeof(float) * (size_t)N_NODES * DIM);
    float*    pool_s1= (float*)carve(sizeof(float) * N_GRAPH * DIM);
    float*    pool_s2= (float*)carve(sizeof(float) * N_GRAPH * DIM);
    float*    pool_t1= (float*)carve(sizeof(float) * N_GRAPH * DIM);
    float*    pool_t2= (float*)carve(sizeof(float) * N_GRAPH * DIM);
    float*    logit_s= (float*)carve(sizeof(float) * N_GRAPH * NCLS);
    float*    logit_t= (float*)carve(sizeof(float) * N_GRAPH * NCLS);
    float*    accum  = (float*)carve(sizeof(float) * 4);

    const int B = 256;
    const long long NF = (long long)N_NODES * DIM;   // node-feature elements

    // ---- degrees -> dinv (self-loop included via init=1.0) ----
    k_fill<<<gdim(N_NODES, B), B, 0, stream>>>(dinv_s, 1.0f, N_NODES);
    k_fill<<<gdim(N_NODES, B), B, 0, stream>>>(dinv_t, 1.0f, N_NODES);
    k_deg_count<<<gdim(N_EDGES, B), B, 0, stream>>>(ei_s + N_EDGES, dinv_s, N_EDGES);
    k_deg_count<<<gdim(N_EDGES, B), B, 0, stream>>>(ei_t + N_EDGES, dinv_t, N_EDGES);
    k_rsqrt_inplace<<<gdim(N_NODES, B), B, 0, stream>>>(dinv_s, N_NODES);
    k_rsqrt_inplace<<<gdim(N_NODES, B), B, 0, stream>>>(dinv_t, N_NODES);

    // ---- graph counts ----
    k_fill<<<gdim(N_GRAPH, B), B, 0, stream>>>(cnt_s, 0.0f, N_GRAPH);
    k_fill<<<gdim(N_GRAPH, B), B, 0, stream>>>(cnt_t, 0.0f, N_GRAPH);
    k_count_batch<<<gdim(N_NODES, B), B, 0, stream>>>(batch_s, cnt_s, N_NODES);
    k_count_batch<<<gdim(N_NODES, B), B, 0, stream>>>(batch_t, cnt_t, N_NODES);

    const float* layW[2]  = { W1, W2 };
    const float* layB[2]  = { b1, b2 };
    float*       pools[2][2] = { { pool_s1, pool_t1 }, { pool_s2, pool_t2 } };
    const int*   ei[2]    = { ei_s, ei_t };
    const int*   batch[2] = { batch_s, batch_t };
    const float* dinv[2]  = { dinv_s, dinv_t };
    float*       agg[2]   = { agg_s, agg_t };
    const float* feat[2]  = { feat_s, feat_t };

    for (int l = 0; l < 2; ++l) {
        // W -> f16, transposed to N-major for contiguous B-fragment LDS reads
        k_w_to_f16_t<<<gdim(DIM * DIM, B), B, 0, stream>>>(layW[l], Wt);
        for (int d = 0; d < 2; ++d) {
            const float* cur = (l == 0) ? feat[d] : agg[d];
            // cast features to f16 for WMMA
            k_f32_to_f16<<<gdim(NF, B), B, 0, stream>>>(cur, xh, (int)NF);
            // H = X @ W  (WMMA, one 16x16 tile per wave)
            k_gemm_wmma<<<MTILES, 256, 0, stream>>>(xh, Wt, hbuf);
            // self-loop + bias init, then edge scatter with atomics
            k_agg_init<<<gdim(NF, B), B, 0, stream>>>(agg[d], hbuf, dinv[d], layB[l], (int)NF);
            k_scatter<<<gdim((long long)N_EDGES * 32, B), B, 0, stream>>>(
                ei[d], ei[d] + N_EDGES, hbuf, dinv[d], agg[d], N_EDGES);
            // mean pool
            k_fill<<<gdim(N_GRAPH * DIM, B), B, 0, stream>>>(pools[l][d], 0.0f, N_GRAPH * DIM);
            k_pool_sum<<<gdim(NF, B), B, 0, stream>>>(agg[d], batch[d], pools[l][d], (int)NF);
            k_pool_div<<<gdim(N_GRAPH * DIM, B), B, 0, stream>>>(pools[l][d],
                (d == 0) ? cnt_s : cnt_t, N_GRAPH * DIM);
        }
    }

    // ---- FC heads ----
    k_fc<<<gdim(N_GRAPH, B), B, 0, stream>>>(pool_s2, fcW1, fcb1, fcW2, fcb2, logit_s, N_GRAPH);
    k_fc<<<gdim(N_GRAPH, B), B, 0, stream>>>(pool_t2, fcW1, fcb1, fcW2, fcb2, logit_t, N_GRAPH);

    // ---- losses ----
    k_fill<<<1, B, 0, stream>>>(accum, 0.0f, 4);
    k_class_loss<<<gdim(N_GRAPH * NCLS, B), B, 0, stream>>>(logit_s, labels, accum);
    k_domain_loss<<<gdim(2 * N_GRAPH, B), B, 0, stream>>>(
        pool_s1, pool_s2, logit_s, pool_t1, pool_t2, logit_t, dW, db, accum);
    k_finalize<<<1, 64, 0, stream>>>(accum, (float*)d_out);
}
